// DeformableAttention_40638980554954
// MI455X (gfx1250) — matrix operand order repaired and checked by reading, not compiled, not run on hardware
//
#include <hip/hip_runtime.h>

// ---------------- static problem config (matches reference) ----------------
#define BATCH 4
#define LQ    21760
#define LV    21760          // 16384 + 4096 + 1024 + 256
#define CDIM  256
#define NHEAD 8
#define NPTS  4
#define HDIM  32             // CDIM / NHEAD == wave32 lanes
#define NLVL  4
#define NQK   96             // 64 offset cols + 32 attn cols (fused proj)

typedef __attribute__((ext_vector_type(16))) __bf16 v16bf;
typedef __attribute__((ext_vector_type(8)))  float  v8f;

// ---------------- prep: fp32 -> bf16 conversions ----------------
__global__ void k_f32_to_bf16(const float* __restrict__ src,
                              __bf16* __restrict__ dst, size_t n) {
  size_t i = (size_t)blockIdx.x * blockDim.x + threadIdx.x;
  size_t stride = (size_t)gridDim.x * blockDim.x;
  for (; i < n; i += stride) dst[i] = (__bf16)src[i];
}

// feat level [B, HW, C] fp32 -> rows [b*LV + start + pos][C] of concat bf16 buffer
__global__ void k_feat_concat_bf16(const float* __restrict__ src,
                                   __bf16* __restrict__ dst,
                                   int HW, int start) {
  size_t n = (size_t)BATCH * HW * CDIM;
  size_t per = (size_t)HW * CDIM;
  size_t i = (size_t)blockIdx.x * blockDim.x + threadIdx.x;
  size_t stride = (size_t)gridDim.x * blockDim.x;
  for (; i < n; i += stride) {
    size_t b = i / per, rem = i % per;
    dst[(b * LV + start) * CDIM + rem] = (__bf16)src[i];
  }
}

// W [K x N] fp32 -> Wt [N x K] bf16 (so B fragments load contiguously)
__global__ void k_transpose_bf16(const float* __restrict__ W,
                                 __bf16* __restrict__ Wt, int K, int N) {
  size_t n = (size_t)K * N;
  size_t i = (size_t)blockIdx.x * blockDim.x + threadIdx.x;
  size_t stride = (size_t)gridDim.x * blockDim.x;
  for (; i < n; i += stride) {
    int nn = (int)(i / K), kk = (int)(i % K);
    Wt[i] = (__bf16)W[(size_t)kk * N + nn];
  }
}

// pack [W_off | W_attn] -> WqkT [96 x 256] bf16, and bias [96] fp32
__global__ void k_pack_qk(const float* __restrict__ Woff,
                          const float* __restrict__ Wattn,
                          const float* __restrict__ boff,
                          const float* __restrict__ battn,
                          __bf16* __restrict__ WqkT, float* __restrict__ bqk) {
  int i = blockIdx.x * blockDim.x + threadIdx.x;   // over 96*256
  if (i < NQK * CDIM) {
    int n = i >> 8, k = i & 255;
    WqkT[i] = (__bf16)(n < 64 ? Woff[(size_t)k * 64 + n]
                              : Wattn[(size_t)k * 32 + (n - 64)]);
  }
  if (i < NQK) bqk[i] = (i < 64) ? boff[i] : battn[i - 64];
}

// ---------------- WMMA GEMM: C[M,N] = A[M,K]bf16 * Bt[N,K]bf16^T + bias ----
// One wave per 16x16 tile; K stepped by 32 via v_wmma_f32_16x16x32_bf16.
template <bool OUT_BF16>
__global__ __launch_bounds__(256)
void k_gemm_wmma(const __bf16* __restrict__ A, const __bf16* __restrict__ Bt,
                 const float* __restrict__ bias,
                 float* __restrict__ Cf, __bf16* __restrict__ Cbf,
                 int M, int N, int K) {
  const int lane  = threadIdx.x & 31;
  const int wave  = threadIdx.x >> 5;
  const int mtile = blockIdx.x;
  const int ntile = blockIdx.y * 8 + wave;
  if (ntile * 16 >= N) return;                 // uniform per-wave: EXEC stays all-1s

  const int l15  = lane & 15;
  const int koff = lane >> 4;                  // half-wave selects K sub-range
  const int mrow = mtile * 16 + l15;           // A-fragment row (16-bit A layout)
  const int ncol = ntile * 16 + l15;           // B-fragment column

  const __bf16* arow = A  + (size_t)mrow * K + koff * 8;   // K = koff*8 + {0..7, 16..23}
  const __bf16* brow = Bt + (size_t)ncol * K + koff * 8;   // K = koff*8 + {0..7, 16..23}

  v8f acc = {};
  for (int kk = 0; kk < K; kk += 32) {
    __builtin_prefetch(arow + kk + 32, 0, 3);  // global_prefetch for next K slab
    union { uint4 u; __bf16 h[8]; } a0, a1, b0, b1;
    a0.u = *(const uint4*)(arow + kk);
    a1.u = *(const uint4*)(arow + kk + 16);
    b0.u = *(const uint4*)(brow + kk);
    b1.u = *(const uint4*)(brow + kk + 16);
    v16bf a, b;
#pragma unroll
    for (int e = 0; e < 8; ++e) {
      a[e] = a0.h[e]; a[8 + e] = a1.h[e];
      b[e] = b0.h[e]; b[8 + e] = b1.h[e];
    }
    acc = __builtin_amdgcn_wmma_f32_16x16x32_bf16(
        /*neg_a=*/false, a, /*neg_b=*/false, b,
        /*c_mod=*/(short)0, acc, /*reuse_a=*/false, /*reuse_b=*/false);
  }

  const float bn = bias ? bias[ncol] : 0.0f;
  const int mbase = mtile * 16 + koff * 8;     // C/D layout: VGPR r -> M = r + 8*halfwave
#pragma unroll
  for (int r = 0; r < 8; ++r) {
    const size_t o = (size_t)(mbase + r) * N + ncol;
    const float v = acc[r] + bn;
    if (OUT_BF16) Cbf[o] = (__bf16)v;
    else          Cf[o]  = v;
  }
}

// ---------------- softmax over NP=4 attn logits (in place, cols 64..95) ----
__global__ void k_softmax_np(float* __restrict__ po) {
  int i = blockIdx.x * blockDim.x + threadIdx.x;     // over B*LQ*NHEAD
  if (i >= BATCH * LQ * NHEAD) return;
  int g = i / NHEAD, h = i % NHEAD;
  float* p = po + (size_t)g * NQK + 64 + h * 4;
  float m = fmaxf(fmaxf(p[0], p[1]), fmaxf(p[2], p[3]));
  float e0 = __expf(p[0] - m), e1 = __expf(p[1] - m);
  float e2 = __expf(p[2] - m), e3 = __expf(p[3] - m);
  float inv = 1.0f / (e0 + e1 + e2 + e3);
  p[0] = e0 * inv; p[1] = e1 * inv; p[2] = e2 * inv; p[3] = e3 * inv;
}

// ---------------- gather + attention-weighted sum ----------------
// one wave per (b,q); lane = channel within head (HDIM == 32)
__global__ __launch_bounds__(256)
void k_gather_attend(const float* __restrict__ refpts,
                     const float* __restrict__ po,
                     const __bf16* __restrict__ value,
                     __bf16* __restrict__ mid) {
  const int lane = threadIdx.x & 31;
  const int gid  = blockIdx.x * 8 + (threadIdx.x >> 5);
  if (gid >= BATCH * LQ) return;
  const int b = gid / LQ;

  const int lvlW[NLVL] = {128, 64, 32, 16};
  const int lvlH[NLVL] = {128, 64, 32, 16};
  const int lvlS[NLVL] = {0, 16384, 20480, 21504};

  const float* pq = po + (size_t)gid * NQK;          // [0..63]=offs, [64..95]=attn
  const float* rq = refpts + (size_t)gid * (NLVL * 2);

  for (int h = 0; h < NHEAD; ++h) {
    float acc = 0.0f;
    for (int lvl = 0; lvl < NLVL; ++lvl) {
      const float rx = rq[lvl * 2 + 0];
      const float ry = rq[lvl * 2 + 1];
      const int Wl = lvlW[lvl], Hl = lvlH[lvl], st = lvlS[lvl];
#pragma unroll
      for (int p = 0; p < NPTS; ++p) {
        const float sx = fminf(fmaxf(rx + pq[h * 8 + p * 2 + 0], 0.0f), 1.0f);
        const float sy = fminf(fmaxf(ry + pq[h * 8 + p * 2 + 1], 0.0f), 1.0f);
        const int x0 = (int)floorf(sx * (float)(Wl - 1));
        const int y0 = (int)floorf(sy * (float)(Hl - 1));
        const int idx = y0 * Wl + x0;
        const __bf16* v =
            value + ((size_t)(b * LV + st + idx) * CDIM) + h * HDIM + lane;
        acc += pq[64 + h * 4 + p] * (float)v[0];
      }
    }
    mid[(size_t)gid * CDIM + h * HDIM + lane] = (__bf16)acc;
  }
}

// ---------------- launch ----------------
extern "C" void kernel_launch(void* const* d_in, const int* in_sizes, int n_in,
                              void* d_out, int out_size, void* d_ws, size_t ws_size,
                              hipStream_t stream) {
  const float* query = (const float*)d_in[0];
  const float* refp  = (const float*)d_in[1];
  const float* feat0 = (const float*)d_in[2];
  const float* feat1 = (const float*)d_in[3];
  const float* feat2 = (const float*)d_in[4];
  const float* feat3 = (const float*)d_in[5];
  const float* W_off  = (const float*)d_in[6];
  const float* b_off  = (const float*)d_in[7];
  const float* W_attn = (const float*)d_in[8];
  const float* b_attn = (const float*)d_in[9];
  const float* W_val  = (const float*)d_in[10];
  const float* b_val  = (const float*)d_in[11];
  const float* W_out  = (const float*)d_in[12];
  const float* b_out  = (const float*)d_in[13];
  float* out = (float*)d_out;

  const size_t ROWS = (size_t)BATCH * LV;       // 87040 == BATCH*LQ too
  const size_t SZ_BF = ROWS * CDIM * sizeof(__bf16);   // 44,564,480 B

  char* ws = (char*)d_ws;
  __bf16* Asrc  = (__bf16*)(ws);                       // concat feats bf16 (reused as mid)
  __bf16* Qbf   = (__bf16*)(ws + SZ_BF);
  __bf16* Vbf   = (__bf16*)(ws + 2 * SZ_BF);           // projected value bf16
  float*  po    = (float*)(ws + 3 * SZ_BF);            // [ROWS, 96] fp32
  char*   tail  = ws + 3 * SZ_BF + ROWS * NQK * sizeof(float);
  __bf16* WvalT = (__bf16*)(tail);
  __bf16* WoutT = (__bf16*)(tail + CDIM * CDIM * sizeof(__bf16));
  __bf16* WqkT  = (__bf16*)(tail + 2 * CDIM * CDIM * sizeof(__bf16));
  float*  bqk   = (float*)(tail + 2 * CDIM * CDIM * sizeof(__bf16)
                                + NQK * CDIM * sizeof(__bf16));
  __bf16* mid   = Asrc;   // alias: concat feats are dead after value projection

  // 1) conversions
  k_f32_to_bf16<<<4096, 256, 0, stream>>>(query, Qbf, ROWS * CDIM);
  k_feat_concat_bf16<<<4096, 256, 0, stream>>>(feat0, Asrc, 128 * 128, 0);
  k_feat_concat_bf16<<<2048, 256, 0, stream>>>(feat1, Asrc, 64 * 64, 16384);
  k_feat_concat_bf16<<<512,  256, 0, stream>>>(feat2, Asrc, 32 * 32, 20480);
  k_feat_concat_bf16<<<128,  256, 0, stream>>>(feat3, Asrc, 16 * 16, 21504);
  k_transpose_bf16<<<256, 256, 0, stream>>>(W_val, WvalT, CDIM, CDIM);
  k_transpose_bf16<<<256, 256, 0, stream>>>(W_out, WoutT, CDIM, CDIM);
  k_pack_qk<<<(NQK * CDIM + 255) / 256, 256, 0, stream>>>(
      W_off, W_attn, b_off, b_attn, WqkT, bqk);

  const int MT = (int)(ROWS / 16);   // 5440

  // 2) value projection -> bf16 (L2-resident for gather)
  k_gemm_wmma<true><<<dim3(MT, 2), 256, 0, stream>>>(
      Asrc, WvalT, b_val, nullptr, Vbf, (int)ROWS, CDIM, CDIM);

  // 3) fused sampling-offset + attention-logit projection -> fp32
  k_gemm_wmma<false><<<dim3(MT, 1), 256, 0, stream>>>(
      Qbf, WqkT, bqk, po, nullptr, (int)ROWS, NQK, CDIM);

  // 4) softmax over the 4 points per head (in place)
  k_softmax_np<<<(BATCH * LQ * NHEAD + 255) / 256, 256, 0, stream>>>(po);

  // 5) deformable gather + weighted sum -> mid (bf16)
  k_gather_attend<<<(BATCH * LQ) / 8, 256, 0, stream>>>(refp, po, Vbf, mid);

  // 6) output projection -> fp32 d_out
  k_gemm_wmma<false><<<dim3(MT, 2), 256, 0, stream>>>(
      mid, WoutT, b_out, out, nullptr, (int)ROWS, CDIM, CDIM);
}
